// Qwen3Attention_34153579938201
// MI455X (gfx1250) — matrix operand order repaired and checked by reading, not compiled, not run on hardware
//
#include <hip/hip_runtime.h>

// ---- problem constants (from the reference) ----
#define BB   2
#define TT   2048
#define HID  1536
#define NH   12
#define NKV  2
#define DD   128

typedef __attribute__((ext_vector_type(8)))  __bf16 v8bf;
typedef __attribute__((ext_vector_type(16))) __bf16 v16bf;
typedef __attribute__((ext_vector_type(8)))  float  v8f;
typedef __attribute__((ext_vector_type(4)))  unsigned int u32x4;
typedef __attribute__((ext_vector_type(8)))  unsigned int u32x8;

__device__ inline unsigned short f2bf(float f) {
    union { float f; unsigned u; } v; v.f = f;
    unsigned r = v.u + 0x7FFFu + ((v.u >> 16) & 1u);
    return (unsigned short)(r >> 16);
}

__device__ inline v8f vzero8() {
    v8f z;
#pragma unroll
    for (int i = 0; i < 8; ++i) z[i] = 0.f;
    return z;
}

// ---------------- TDM: 2D bf16 tile (tile_d0 x tile_d1) global -> LDS ----------------
// Builds the D# descriptor (CDNA5 ISA) in SGPR tuples and
// issues tensor_load_to_lds. Tracked with TENSORcnt (in-order per wave).
__device__ inline void tdm_load_2d(unsigned lds_addr, const unsigned short* gptr,
                                   unsigned tensor_d0, unsigned tensor_d1,
                                   unsigned tile_d0, unsigned tile_d1,
                                   unsigned stride_elems) {
    unsigned long long ga = (unsigned long long)(size_t)gptr;
    u32x4 g0;
    g0[0] = 1u;                                   // count=1 (valid user descriptor)
    g0[1] = lds_addr;                             // LDS byte address
    g0[2] = (unsigned)(ga & 0xFFFFFFFFu);         // global_addr[31:0]
    g0[3] = (unsigned)((ga >> 32) & 0x1FFFFFFu)   // global_addr[56:32]
          | (2u << 30);                           // type = 2 ("image")
    u32x8 g1;
    g1[0] = (1u << 16);                           // data_size=1 -> 2 bytes/elem
    g1[1] = (tensor_d0 & 0xFFFFu) << 16;          // tensor_dim0[15:0]
    g1[2] = (tensor_d0 >> 16) | ((tensor_d1 & 0xFFFFu) << 16);
    g1[3] = (tensor_d1 >> 16) | (tile_d0 << 16);  // tensor_dim1[31:16] | tile_dim0
    g1[4] = tile_d1;                              // tile_dim1 (tile_dim2 = 0)
    g1[5] = stride_elems;                         // tensor_dim0_stride[31:0]
    g1[6] = 0;                                    // stride[47:32] | dim1_stride[15:0]
    g1[7] = 0;
    u32x4 gz;
    gz[0] = 0; gz[1] = 0; gz[2] = 0; gz[3] = 0;   // groups 2/3: 2D tile, unused dims
    asm volatile("tensor_load_to_lds %0, %1, %2, %3"
                 :: "s"(g0), "s"(g1), "s"(gz), "s"(gz)
                 : "memory");
}

// ---------------- f32 -> bf16 convert ----------------
__global__ void cvt_bf16_kernel(const float* __restrict__ src,
                                unsigned short* __restrict__ dst, int n) {
    int i = blockIdx.x * blockDim.x + threadIdx.x;
    int stride = gridDim.x * blockDim.x;
    for (; i < n; i += stride) dst[i] = f2bf(src[i]);
}

// ---------------- bf16 GEMM: Out[M,N] = A[M,K] * W[N,K]^T ----------------
// Block = 256 threads (8 waves), tile 128(M) x 64(N), BK = 32, TDM double-buffered LDS.
// Wave (wm,wn) computes a 32x32 sub-tile = 2x2 WMMA 16x16 tiles.
template<int N, int K>
__global__ __launch_bounds__(256, 1)
void gemm_bf16_kernel(const unsigned short* __restrict__ A,
                      const unsigned short* __restrict__ W,
                      float* __restrict__ Out, int M) {
    __shared__ __attribute__((aligned(64))) unsigned short lds_a[2][128 * 32];
    __shared__ __attribute__((aligned(64))) unsigned short lds_b[2][64 * 32];

    const int tid  = threadIdx.x;
    const int lane = tid & 31;
    const int wave = tid >> 5;
    const int ln   = lane & 15;
    const int g    = lane >> 4;
    const int wm   = wave & 3;      // 0..3  -> M sub-block of 32
    const int wn   = wave >> 2;     // 0..1  -> N sub-block of 32
    const int m0   = blockIdx.x * 128;
    const int n0   = blockIdx.y * 64;
    const int nsteps = K / 32;

    v8f acc[2][2];
#pragma unroll
    for (int i = 0; i < 2; ++i)
#pragma unroll
        for (int j = 0; j < 2; ++j) acc[i][j] = vzero8();

    // prologue: DMA first K-slice into buffer 0 (wave 0 drives the TDM)
    if (wave == 0) {
        tdm_load_2d((unsigned)(size_t)&lds_a[0][0], A + (size_t)m0 * K,
                    (unsigned)K, (unsigned)M, 32u, 128u, (unsigned)K);
        tdm_load_2d((unsigned)(size_t)&lds_b[0][0], W + (size_t)n0 * K,
                    (unsigned)K, (unsigned)N, 32u, 64u, (unsigned)K);
    }

    for (int i = 0; i < nsteps; ++i) {
        const int cur = i & 1;
        if (wave == 0) {
            if (i + 1 < nsteps) {
                const int k0n = (i + 1) * 32;
                tdm_load_2d((unsigned)(size_t)&lds_a[1 - cur][0],
                            A + (size_t)m0 * K + k0n,
                            (unsigned)K, (unsigned)M, 32u, 128u, (unsigned)K);
                tdm_load_2d((unsigned)(size_t)&lds_b[1 - cur][0],
                            W + (size_t)n0 * K + k0n,
                            (unsigned)K, (unsigned)N, 32u, 64u, (unsigned)K);
                // TDM ops complete in-order: cnt<=2 => current buffer is resident
                __builtin_amdgcn_s_wait_tensorcnt(2);
            } else {
                __builtin_amdgcn_s_wait_tensorcnt(0);
            }
        }
        __syncthreads();

        v16bf af[2], bfr[2];
#pragma unroll
        for (int mf = 0; mf < 2; ++mf) {
            const int row = wm * 32 + mf * 16 + ln;
            const int c0  = g * 16;
            const unsigned short* p = &lds_a[cur][row * 32];
            v8bf a0 = *(const v8bf*)(p + c0);
            v8bf a1 = *(const v8bf*)(p + c0 + 8);
#pragma unroll
            for (int q = 0; q < 8; ++q) { af[mf][q] = a0[q]; af[mf][8 + q] = a1[q]; }
        }
#pragma unroll
        for (int nf = 0; nf < 2; ++nf) {
            const int n = wn * 32 + nf * 16 + ln;
            bfr[nf] = *(const v16bf*)&lds_b[cur][n * 32 + g * 16];
        }
#pragma unroll
        for (int mf = 0; mf < 2; ++mf)
#pragma unroll
            for (int nf = 0; nf < 2; ++nf)
                acc[mf][nf] = __builtin_amdgcn_wmma_f32_16x16x32_bf16(
                    false, af[mf], false, bfr[nf], (short)0, acc[mf][nf], false, false);
        __syncthreads();   // all waves done reading before next DMA overwrites
    }

#pragma unroll
    for (int mf = 0; mf < 2; ++mf)
#pragma unroll
        for (int nf = 0; nf < 2; ++nf)
#pragma unroll
            for (int r = 0; r < 8; ++r) {
                const int mm = m0 + wm * 32 + mf * 16 + r + 8 * g;
                const int nn = n0 + wn * 32 + nf * 16 + ln;
                Out[(size_t)mm * N + nn] = acc[mf][nf][r];
            }
}

// ---------------- RMS-norm + RoPE -> bf16, permute (b,t,h,d) -> (b,h,t,d) ----------------
__global__ void norm_rope_kernel(const float* __restrict__ src,
                                 const float* __restrict__ nw,
                                 const float* __restrict__ cosb,
                                 const float* __restrict__ sinb,
                                 unsigned short* __restrict__ dst, int NHloc) {
    const int idx = blockIdx.x;
    const int hh = idx % NHloc;
    const int bt = idx / NHloc;
    const int t  = bt % TT;
    const int b  = bt / TT;
    const int d  = threadIdx.x;

    __shared__ float sh[DD];
    const float x = src[((size_t)(b * TT + t) * NHloc + hh) * DD + d];
    sh[d] = x * x;
    __syncthreads();
#pragma unroll
    for (int off = 64; off > 0; off >>= 1) {
        if (d < off) sh[d] += sh[d + off];
        __syncthreads();
    }
    const float var = sh[0] * (1.f / DD);
    const float inv = rsqrtf(var + 1e-6f);
    const float nv  = x * inv * nw[d];
    __syncthreads();
    sh[d] = nv;
    __syncthreads();
    const float rot = (d < 64) ? -sh[d + 64] : sh[d - 64];
    const float c = cosb[((size_t)b * TT + t) * DD + d];
    const float s = sinb[((size_t)b * TT + t) * DD + d];
    const float o = nv * c + rot * s;
    dst[(((size_t)b * NHloc + hh) * TT + t) * DD + d] = f2bf(o);
}

// ---------------- V (b,t,hkv,d) f32 -> V^T (b,hkv,d,t) bf16 ----------------
__global__ void vtrans_kernel(const float* __restrict__ src,
                              unsigned short* __restrict__ dst) {
    const int idx = blockIdx.x;
    const int hh = idx % NKV;
    const int bt = idx / NKV;
    const int t  = bt % TT;
    const int b  = bt / TT;
    const int d  = threadIdx.x;
    const float x = src[((size_t)(b * TT + t) * NKV + hh) * DD + d];
    dst[(((size_t)b * NKV + hh) * DD + d) * TT + t] = f2bf(x);
}

// ---------------- Flash attention (causal), bf16 WMMA, f32 softmax ----------------
// grid.x = B*H*(T/128); 8 waves per block, each wave owns 16 query rows.
// K (32x128) and V^T (128x32) tiles are staged once per block via TDM, double-buffered.
__global__ __launch_bounds__(256, 1)
void flash_attn_kernel(const unsigned short* __restrict__ qb,
                       const unsigned short* __restrict__ kbuf,
                       const unsigned short* __restrict__ vtb,
                       unsigned short* __restrict__ ao) {
    const float scale = 0.08838834764831845f; // D^-0.5
    const int blk  = blockIdx.x;
    const int qblk = blk & 15;            // T/128 = 16
    const int bh   = blk >> 4;
    const int h    = bh % NH;
    const int b    = bh / NH;
    const int hkv  = h / (NH / NKV);
    const int lane = threadIdx.x & 31;
    const int wave = threadIdx.x >> 5;
    const int ln   = lane & 15;
    const int g    = lane >> 4;
    const int qbase = qblk * 128 + wave * 16;

    const unsigned short* Q  = qb   + (size_t)(b * NH  + h)   * TT * DD;
    const unsigned short* K  = kbuf + (size_t)(b * NKV + hkv) * TT * DD;
    const unsigned short* VT = vtb  + (size_t)(b * NKV + hkv) * DD * TT;

    // LDS staging: K tile [32 keys][128 d], V tile [128 d][32 keys], double-buffered
    __shared__ __attribute__((aligned(64))) unsigned short kls[2][32 * 128];
    __shared__ __attribute__((aligned(64))) unsigned short vls[2][128 * 32];
    // per-wave P staging buffer (16 x 32 bf16)
    __shared__ __attribute__((aligned(32))) unsigned short pbuf[8][16 * 32];

    // Q fragments: wave's 16 rows x 128 cols as 4 A-fragments (16x32 each)
    v16bf qf[4];
    {
        const unsigned short* qr = Q + (size_t)(qbase + ln) * DD;
#pragma unroll
        for (int kk = 0; kk < 4; ++kk) {
            const int c0 = kk * 32 + g * 16;
            v8bf a0 = *(const v8bf*)(qr + c0);
            v8bf a1 = *(const v8bf*)(qr + c0 + 8);
#pragma unroll
            for (int i = 0; i < 8; ++i) { qf[kk][i] = a0[i]; qf[kk][8 + i] = a1[i]; }
        }
    }

    v8f acc[8];
#pragma unroll
    for (int dt = 0; dt < 8; ++dt) acc[dt] = vzero8();
    float mrow[8], lrow[8];
#pragma unroll
    for (int r = 0; r < 8; ++r) { mrow[r] = -1e30f; lrow[r] = 0.f; }

    const int kendw = (qbase + 16 + 31) & ~31;  // per-wave causal limit (multiple of 32)
    const int nkb   = (qblk + 1) * 4;           // block-uniform key-block count (32 keys each)

    if (wave == 0) {
        tdm_load_2d((unsigned)(size_t)&kls[0][0], K,  DD, TT, 128u, 32u, DD);
        tdm_load_2d((unsigned)(size_t)&vls[0][0], VT, TT, DD, 32u, 128u, TT);
    }

    for (int i = 0; i < nkb; ++i) {
        const int kb  = i * 32;
        const int cur = i & 1;
        if (wave == 0) {
            if (i + 1 < nkb) {
                tdm_load_2d((unsigned)(size_t)&kls[1 - cur][0],
                            K + (size_t)(kb + 32) * DD, DD, TT, 128u, 32u, DD);
                tdm_load_2d((unsigned)(size_t)&vls[1 - cur][0],
                            VT + (kb + 32), TT, DD, 32u, 128u, TT);
                __builtin_amdgcn_s_wait_tensorcnt(2);
            } else {
                __builtin_amdgcn_s_wait_tensorcnt(0);
            }
        }
        __syncthreads();

        if (kb < kendw) {
            // ---- S = Q K^T for 32 keys (two 16x16 tiles), K from LDS ----
            v8f S[2]; S[0] = vzero8(); S[1] = vzero8();
#pragma unroll
            for (int tile = 0; tile < 2; ++tile) {
                const unsigned short* kr = &kls[cur][(tile * 16 + ln) * 128];
#pragma unroll
                for (int kk = 0; kk < 4; ++kk) {
                    v16bf bf = *(const v16bf*)(kr + kk * 32 + g * 16);
                    S[tile] = __builtin_amdgcn_wmma_f32_16x16x32_bf16(
                        false, qf[kk], false, bf, (short)0, S[tile], false, false);
                }
            }
            // ---- scale, causal mask, online softmax ----
            float mnew[8];
#pragma unroll
            for (int r = 0; r < 8; ++r) {
                const int qrow = qbase + r + 8 * g;
                float s0 = S[0][r] * scale;
                float s1 = S[1][r] * scale;
                if (kb + ln > qrow)      s0 = -1e30f;
                if (kb + 16 + ln > qrow) s1 = -1e30f;
                S[0][r] = s0; S[1][r] = s1;
                float mx = fmaxf(s0, s1);
#pragma unroll
                for (int off = 1; off < 16; off <<= 1) mx = fmaxf(mx, __shfl_xor(mx, off, 16));
                mnew[r] = fmaxf(mrow[r], mx);
            }
#pragma unroll
            for (int r = 0; r < 8; ++r) {
                const float alpha = __expf(mrow[r] - mnew[r]);
                const float p0 = __expf(S[0][r] - mnew[r]);
                const float p1 = __expf(S[1][r] - mnew[r]);
                float ps = p0 + p1;
#pragma unroll
                for (int off = 1; off < 16; off <<= 1) ps += __shfl_xor(ps, off, 16);
                lrow[r] = lrow[r] * alpha + ps;
                mrow[r] = mnew[r];
#pragma unroll
                for (int dt = 0; dt < 8; ++dt) acc[dt][r] = acc[dt][r] * alpha;
                const int m = r + 8 * g;
                pbuf[wave][m * 32 + ln]      = f2bf(p0);
                pbuf[wave][m * 32 + 16 + ln] = f2bf(p1);
            }
            // ---- P (16x32) as A-fragment from LDS ----
            v16bf pf;
            {
                const unsigned short* pr = &pbuf[wave][ln * 32];
                const int c0 = g * 16;
                v8bf a0 = *(const v8bf*)(pr + c0);
                v8bf a1 = *(const v8bf*)(pr + c0 + 8);
#pragma unroll
                for (int i2 = 0; i2 < 8; ++i2) { pf[i2] = a0[i2]; pf[8 + i2] = a1[i2]; }
            }
            // ---- O += P * V, V from LDS ----
#pragma unroll
            for (int dt = 0; dt < 8; ++dt) {
                v16bf vf = *(const v16bf*)&vls[cur][(dt * 16 + ln) * 32 + g * 16];
                acc[dt] = __builtin_amdgcn_wmma_f32_16x16x32_bf16(
                    false, pf, false, vf, (short)0, acc[dt], false, false);
            }
        }
        __syncthreads();
    }

    // ---- normalize, write attn-out bf16 (b, t, h*D) ----
#pragma unroll
    for (int r = 0; r < 8; ++r) {
        const float inv = 1.f / lrow[r];
        const int t = qbase + r + 8 * g;
        unsigned short* orow = ao + ((size_t)b * TT + t) * (NH * DD) + h * DD;
#pragma unroll
        for (int dt = 0; dt < 8; ++dt)
            orow[dt * 16 + ln] = f2bf(acc[dt][r] * inv);
    }
}

// ---------------- launch ----------------
extern "C" void kernel_launch(void* const* d_in, const int* in_sizes, int n_in,
                              void* d_out, int out_size, void* d_ws, size_t ws_size,
                              hipStream_t stream) {
    (void)in_sizes; (void)n_in; (void)out_size; (void)ws_size;
    const float* hs   = (const float*)d_in[0];
    const float* cosb = (const float*)d_in[1];
    const float* sinb = (const float*)d_in[2];
    // d_in[3] = attention_mask: pure causal mask, reconstructed analytically
    const float* q_w  = (const float*)d_in[4];
    const float* k_w  = (const float*)d_in[5];
    const float* v_w  = (const float*)d_in[6];
    const float* o_w  = (const float*)d_in[7];
    const float* qnw  = (const float*)d_in[8];
    const float* knw  = (const float*)d_in[9];
    float* out = (float*)d_out;

    char* ws = (char*)d_ws;
    size_t off = 0;
    auto alloc = [&](size_t bytes) -> void* {
        off = (off + 255) & ~(size_t)255;
        void* p = ws + off;
        off += bytes;
        return p;
    };

    const size_t MT = (size_t)BB * TT;                 // 4096
    unsigned short* hs_bf = (unsigned short*)alloc(MT * HID * 2);
    unsigned short* qw_bf = (unsigned short*)alloc((size_t)(NH  * DD) * HID * 2);
    unsigned short* kw_bf = (unsigned short*)alloc((size_t)(NKV * DD) * HID * 2);
    unsigned short* vw_bf = (unsigned short*)alloc((size_t)(NKV * DD) * HID * 2);
    unsigned short* ow_bf = (unsigned short*)alloc((size_t)HID * (NH * DD) * 2);
    float* q_raw = (float*)alloc(MT * (NH  * DD) * 4);
    float* k_raw = (float*)alloc(MT * (NKV * DD) * 4);
    float* v_raw = (float*)alloc(MT * (NKV * DD) * 4);
    unsigned short* q_bf  = (unsigned short*)alloc(MT * (NH  * DD) * 2);
    unsigned short* k_bf  = (unsigned short*)alloc(MT * (NKV * DD) * 2);
    unsigned short* vt_bf = (unsigned short*)alloc(MT * (NKV * DD) * 2);
    unsigned short* ao_bf = (unsigned short*)alloc(MT * (NH  * DD) * 2);

    // 1) f32 -> bf16 converts
    cvt_bf16_kernel<<<1024, 256, 0, stream>>>(hs,  hs_bf, (int)(MT * HID));
    cvt_bf16_kernel<<<1024, 256, 0, stream>>>(q_w, qw_bf, NH  * DD * HID);
    cvt_bf16_kernel<<<512,  256, 0, stream>>>(k_w, kw_bf, NKV * DD * HID);
    cvt_bf16_kernel<<<512,  256, 0, stream>>>(v_w, vw_bf, NKV * DD * HID);
    cvt_bf16_kernel<<<1024, 256, 0, stream>>>(o_w, ow_bf, HID * NH * DD);

    // 2) QKV projections (WMMA bf16, f32 accum, TDM double-buffered)
    gemm_bf16_kernel<NH * DD, HID><<<dim3(32, 24), 256, 0, stream>>>(hs_bf, qw_bf, q_raw, (int)MT);
    gemm_bf16_kernel<NKV * DD, HID><<<dim3(32, 4), 256, 0, stream>>>(hs_bf, kw_bf, k_raw, (int)MT);
    gemm_bf16_kernel<NKV * DD, HID><<<dim3(32, 4), 256, 0, stream>>>(hs_bf, vw_bf, v_raw, (int)MT);

    // 3) RMS-norm + RoPE -> bf16 (b,h,t,d); V -> bf16 transposed (b,hkv,d,t)
    norm_rope_kernel<<<BB * TT * NH,  DD, 0, stream>>>(q_raw, qnw, cosb, sinb, q_bf, NH);
    norm_rope_kernel<<<BB * TT * NKV, DD, 0, stream>>>(k_raw, knw, cosb, sinb, k_bf, NKV);
    vtrans_kernel<<<BB * TT * NKV, DD, 0, stream>>>(v_raw, vt_bf);

    // 4) flash attention (TDM-staged K/V tiles)
    flash_attn_kernel<<<BB * NH * (TT / 128), 256, 0, stream>>>(q_bf, k_bf, vt_bf, ao_bf);

    // 5) output projection -> f32 d_out
    gemm_bf16_kernel<HID, NH * DD><<<dim3(32, 24), 256, 0, stream>>>(ao_bf, ow_bf, out, (int)MT);
}
